// AttentionHead_30837865185841
// MI455X (gfx1250) — matrix operand order repaired and checked by reading, not compiled, not run on hardware
//
#include <hip/hip_runtime.h>
#include <hip/hip_bf16.h>

// Shapes from the reference
#define B_ 4
#define S_ 4096
#define E_ 1024
#define D_ 128

typedef _Float16 v16h __attribute__((ext_vector_type(16)));
typedef _Float16 v8h  __attribute__((ext_vector_type(8)));
typedef float    v8f  __attribute__((ext_vector_type(8)));
typedef unsigned int v4u  __attribute__((ext_vector_type(4)));
typedef int          v4i  __attribute__((ext_vector_type(4)));
typedef int          v8i_ __attribute__((ext_vector_type(8)));

__device__ __forceinline__ v8f wmma_f16(v16h a, v16h b, v8f c) {
    return __builtin_amdgcn_wmma_f32_16x16x32_f16(false, a, false, b,
                                                  (short)0, c, false, false);
}

__device__ __forceinline__ v16h join8(v8h lo, v8h hi) {
    return __builtin_shufflevector(lo, hi, 0, 1, 2, 3, 4, 5, 6, 7,
                                           8, 9, 10, 11, 12, 13, 14, 15);
}

// ---------------------------------------------------------------------------
// TDM: issue a 2D f16 tile load (tile_d0 x tile_d1, dim0 contiguous) into LDS.
// Descriptor per CDNA5 ISA §8.3/8.4 (D# group0 + group1; 2D -> groups 2/3
// carry tensor_dim2=1 so the z index is in-bounds). This toolchain's builtin
// is the 6-arg form: (g0 v4u, g1 v8i, g2 v4i, g3 v4i, extra v8i, cpol i32).
// ---------------------------------------------------------------------------
__device__ __forceinline__ void tdm_load_2d_f16(const _Float16* gsrc,
                                                const _Float16* lds_dst,
                                                unsigned tensor_d0,
                                                unsigned tensor_d1,
                                                unsigned stride0,
                                                unsigned tile_d0,
                                                unsigned tile_d1)
{
    unsigned long long ga = (unsigned long long)(uintptr_t)gsrc;
    unsigned lds_off = (unsigned)(uintptr_t)lds_dst;   // low 32 bits = LDS byte addr

    v4u g0;
    g0[0] = 1u;                                   // count=1, user descriptor
    g0[1] = lds_off;                              // lds_addr
    g0[2] = (unsigned)ga;                         // global_addr[31:0]
    g0[3] = (unsigned)((ga >> 32) & 0x1FFFFFFu)   // global_addr[56:32]
          | (2u << 30);                           // type = 2 ("image")

    v8i_ g1;
    g1[0] = (int)(1u << 16);                      // data_size = 1 (2 bytes)
    g1[1] = (int)((tensor_d0 & 0xFFFFu) << 16);   // tensor_dim0[15:0] @ bits 63:48
    g1[2] = (int)((tensor_d0 >> 16) | ((tensor_d1 & 0xFFFFu) << 16));
    g1[3] = (int)((tensor_d1 >> 16) | (tile_d0 << 16));
    g1[4] = (int)(tile_d1 & 0xFFFFu);             // tile_dim1; tile_dim2 = 0
    g1[5] = (int)stride0;                         // tensor_dim0_stride[31:0]
    g1[6] = 0;                                    // stride hi + dim1_stride lo
    g1[7] = 0;

    v4i g2;                                       // tensor_dim2 = 1 (keep z=0 in range)
    g2[0] = 1; g2[1] = 0; g2[2] = 0; g2[3] = 0;
    v4i g3;
    g3[0] = 0; g3[1] = 0; g3[2] = 0; g3[3] = 0;
    v8i_ gx;                                      // unused trailing group: zeros
    gx[0] = 0; gx[1] = 0; gx[2] = 0; gx[3] = 0;
    gx[4] = 0; gx[5] = 0; gx[6] = 0; gx[7] = 0;

    __builtin_amdgcn_tensor_load_to_lds(g0, g1, g2, g3, gx, 0);
}

// ---------------------------------------------------------------------------
// Kernel 0: Wt[mat][d][e] = (f16) W_mat[e][d]  (tiny one-shot transpose)
// ---------------------------------------------------------------------------
__global__ __launch_bounds__(256)
void wprep_kernel(const float* __restrict__ Wq, const float* __restrict__ Wk,
                  const float* __restrict__ Wv, _Float16* __restrict__ Wt)
{
    int idx = blockIdx.x * 256 + threadIdx.x;     // 3 * D_ * E_ total
    int mat = idx / (D_ * E_);
    int rem = idx % (D_ * E_);
    int d = rem / E_, e = rem % E_;
    const float* W = (mat == 0) ? Wq : (mat == 1) ? Wk : Wv;
    Wt[idx] = (_Float16)W[(size_t)e * D_ + d];
}

// ---------------------------------------------------------------------------
// Kernel 1: q/k/v = x @ W + b (WMMA f16, f32 accum).
// grid = (B*S/16, 3), block 256 (8 waves; wave owns a 16-col N tile).
// mat 0 -> Qw [b][s][d]; mat 1 -> Kw [b][s][d]; mat 2 -> Vtw [b][d][s].
// ---------------------------------------------------------------------------
__global__ __launch_bounds__(256)
void qkv_proj_kernel(const float* __restrict__ x,
                     const _Float16* __restrict__ Wt,
                     const float* __restrict__ bq, const float* __restrict__ bk,
                     const float* __restrict__ bv,
                     _Float16* __restrict__ Qw,
                     _Float16* __restrict__ Kw,
                     _Float16* __restrict__ Vtw)
{
    const int mat = blockIdx.y;
    const float* bias = (mat == 0) ? bq : (mat == 1) ? bk : bv;
    const _Float16* Wm = Wt + (size_t)mat * D_ * E_;   // [d][e]

    const int tiles_per_b = S_ / 16;
    const int b    = blockIdx.x / tiles_per_b;
    const int m0   = (blockIdx.x % tiles_per_b) * 16;
    const int lane = threadIdx.x & 31;
    const int wv   = threadIdx.x >> 5;
    const int n0   = wv * 16;

    const int Mrow  = lane & 15;
    const int hi    = lane >> 4;
    const int koffA = hi * 16;     // A elems 0..15 -> K+koffA
    const int kbB   = hi * 16;     // B elems 0..15 -> K+kbB
    const int nn    = lane & 15;

    const float*    xrow = x  + ((size_t)b * S_ + (m0 + Mrow)) * E_;
    const _Float16* wrow = Wm + (size_t)(n0 + nn) * E_;          // contiguous in e

    v8f c = {};
    for (int e = 0; e < E_; e += 32) {
        v16h a;
#pragma unroll
        for (int i = 0; i < 8; ++i) {
            a[i]     = (_Float16)xrow[e + koffA + i];
            a[i + 8] = (_Float16)xrow[e + koffA + 8 + i];
        }
        v16h bm = *(const v16h*)&wrow[e + kbB];                  // 32B contiguous
        c = wmma_f16(a, bm, c);
    }

    const float badd = bias[n0 + nn];
#pragma unroll
    for (int r = 0; r < 8; ++r) c[r] += badd;

    if (mat == 2) {               // V, transposed [b][d][s]
#pragma unroll
        for (int r = 0; r < 8; ++r) {
            const int row = m0 + r + 8 * hi, col = n0 + nn;
            Vtw[(size_t)b * D_ * S_ + (size_t)col * S_ + row] = (_Float16)c[r];
        }
    } else {
        _Float16* o = (mat == 0) ? Qw : Kw;
#pragma unroll
        for (int r = 0; r < 8; ++r) {
            const int row = m0 + r + 8 * hi, col = n0 + nn;
            o[((size_t)b * S_ + row) * D_ + col] = (_Float16)c[r];
        }
    }
}

// ---------------------------------------------------------------------------
// Kernel 2: flash attention. grid = B*S/64, block 128 (4 waves x 16 q rows).
// K/V tiles staged by the Tensor Data Mover, double-buffered in LDS.
// ---------------------------------------------------------------------------
#define BQW 4
#define BQ  (BQW * 16)
#define BK  32
#define NT  (S_ / BK)

__global__ __launch_bounds__(128)
void flash_attn_kernel(const _Float16* __restrict__ Qw,
                       const _Float16* __restrict__ Kw,
                       const _Float16* __restrict__ Vtw,
                       float* __restrict__ out)
{
    __shared__ __attribute__((aligned(64))) _Float16 ldsK [2][BK * D_];  // [key][d]
    __shared__ __attribute__((aligned(64))) _Float16 ldsVt[2][D_ * BK];  // [d][key]
    __shared__ __attribute__((aligned(64))) _Float16 ldsP [BQW][16 * BK];

    const int tiles_per_b = S_ / BQ;
    const int b    = blockIdx.x / tiles_per_b;
    const int q0   = (blockIdx.x % tiles_per_b) * BQ;
    const int tid  = threadIdx.x;
    const int lane = tid & 31;
    const int wv   = tid >> 5;

    const int Mrow  = lane & 15;
    const int hi    = lane >> 4;
    const int koffA = hi * 16;
    const int kbB   = hi * 16;
    const int nn    = lane & 15;

    const _Float16* KwB  = Kw  + (size_t)b * S_ * D_;   // [s][d]
    const _Float16* VtB  = Vtw + (size_t)b * D_ * S_;   // [d][s]

    // Q A-fragments in registers for the whole kernel (4 K-steps of D=128)
    v16h qa[4];
    {
        const _Float16* qp = Qw + ((size_t)b * S_ + (q0 + wv * 16 + Mrow)) * D_;
#pragma unroll
        for (int ks = 0; ks < 4; ++ks) {
            v8h lo = *(const v8h*)&qp[ks * 32 + koffA];
            v8h hi8 = *(const v8h*)&qp[ks * 32 + koffA + 8];
            qa[ks] = join8(lo, hi8);
        }
    }

    v8f o[8];
#pragma unroll
    for (int t = 0; t < 8; ++t) o[t] = (v8f){};
    float mrow[8], lrow[8];
#pragma unroll
    for (int r = 0; r < 8; ++r) { mrow[r] = -__builtin_inff(); lrow[r] = 0.0f; }

    const float scale = 0.08838834764831845f; // 1/sqrt(128)

    // Prologue: TDM-issue tile 0 into buffer 0 (wave 0 only; TDM ignores EXEC)
    if (wv == 0) {
        tdm_load_2d_f16(KwB, &ldsK[0][0],  D_, S_, D_, D_, BK);
        tdm_load_2d_f16(VtB, &ldsVt[0][0], S_, D_, S_, BK, D_);
    }

    for (int it = 0; it < NT; ++it) {
        if (wv == 0) __builtin_amdgcn_s_wait_tensorcnt(0);  // tile `it` landed
        __syncthreads();                                    // publish to all waves
        const int cur = it & 1;
        if (wv == 0 && it + 1 < NT) {                       // prefetch next tile
            const int kv = (it + 1) * BK;
            tdm_load_2d_f16(KwB + (size_t)kv * D_, &ldsK[cur ^ 1][0],
                            D_, S_, D_, D_, BK);
            tdm_load_2d_f16(VtB + kv,              &ldsVt[cur ^ 1][0],
                            S_, D_, S_, BK, D_);
        }
        const _Float16* Kt_ = &ldsK[cur][0];
        const _Float16* Vt_ = &ldsVt[cur][0];

        // S = Q K^T : two 16-key tiles; B-frag = 32B contiguous per lane
        v8f s0 = (v8f){}, s1 = (v8f){};
#pragma unroll
        for (int ks = 0; ks < 4; ++ks) {
            v16h b0 = *(const v16h*)&Kt_[(size_t)nn * D_ + ks * 32 + kbB];
            v16h b1 = *(const v16h*)&Kt_[(size_t)(16 + nn) * D_ + ks * 32 + kbB];
            s0 = wmma_f16(qa[ks], b0, s0);
            s1 = wmma_f16(qa[ks], b1, s1);
        }

        // Online softmax: row reductions stay inside the 16-lane column groups
#pragma unroll
        for (int r = 0; r < 8; ++r) {
            const float v0 = s0[r] * scale, v1 = s1[r] * scale;
            float rmax = fmaxf(v0, v1);
            rmax = fmaxf(rmax, __shfl_xor(rmax, 1, 32));
            rmax = fmaxf(rmax, __shfl_xor(rmax, 2, 32));
            rmax = fmaxf(rmax, __shfl_xor(rmax, 4, 32));
            rmax = fmaxf(rmax, __shfl_xor(rmax, 8, 32));
            const float mnew  = fmaxf(mrow[r], rmax);
            const float alpha = __expf(mrow[r] - mnew);
            const float p0 = __expf(v0 - mnew);
            const float p1 = __expf(v1 - mnew);
            float rsum = p0 + p1;
            rsum += __shfl_xor(rsum, 1, 32);
            rsum += __shfl_xor(rsum, 2, 32);
            rsum += __shfl_xor(rsum, 4, 32);
            rsum += __shfl_xor(rsum, 8, 32);
            lrow[r] = lrow[r] * alpha + rsum;
            mrow[r] = mnew;
#pragma unroll
            for (int t = 0; t < 8; ++t) o[t][r] *= alpha;
            const int row = r + 8 * hi;
            ldsP[wv][row * BK + nn]      = (_Float16)p0;
            ldsP[wv][row * BK + 16 + nn] = (_Float16)p1;
        }
        // Same-wave LDS RAW (cross-lane): DS is in-order per wave; stop compiler
        // reordering and drain DScnt before re-reading P.
        asm volatile("s_wait_dscnt 0" ::: "memory");

        // P as A-fragment over the 32 keys (two contiguous 16B chunks)
        v8h plo = *(const v8h*)&ldsP[wv][Mrow * BK + koffA];
        v8h phi = *(const v8h*)&ldsP[wv][Mrow * BK + koffA + 8];
        v16h pa = join8(plo, phi);

        // O += P V : B-frag from ldsVt[d][key], 32B contiguous per lane
#pragma unroll
        for (int t = 0; t < 8; ++t) {
            v16h bvf = *(const v16h*)&Vt_[(size_t)(t * 16 + nn) * BK + kbB];
            o[t] = wmma_f16(pa, bvf, o[t]);
        }
    }

    // Epilogue: out = O / l (f32)
#pragma unroll
    for (int r = 0; r < 8; ++r) {
        const float inv = 1.0f / lrow[r];
        const int row = q0 + wv * 16 + r + 8 * hi;
#pragma unroll
        for (int t = 0; t < 8; ++t)
            out[((size_t)b * S_ + row) * D_ + t * 16 + nn] = o[t][r] * inv;
    }
}

// ---------------------------------------------------------------------------
extern "C" void kernel_launch(void* const* d_in, const int* in_sizes, int n_in,
                              void* d_out, int out_size, void* d_ws, size_t ws_size,
                              hipStream_t stream) {
    (void)in_sizes; (void)n_in; (void)out_size; (void)ws_size;
    const float* x  = (const float*)d_in[0];
    const float* Wq = (const float*)d_in[1];
    const float* bq = (const float*)d_in[2];
    const float* Wk = (const float*)d_in[3];
    const float* bk = (const float*)d_in[4];
    const float* Wv = (const float*)d_in[5];
    const float* bv = (const float*)d_in[6];
    float* out = (float*)d_out;

    // Workspace: Q (4MB) + K (4MB) + Vt (4MB) + Wt (0.75MB)
    _Float16* Qw  = (_Float16*)d_ws;
    _Float16* Kw  = Qw  + (size_t)B_ * S_ * D_;
    _Float16* Vtw = Kw  + (size_t)B_ * S_ * D_;
    _Float16* Wt  = Vtw + (size_t)B_ * D_ * S_;

    wprep_kernel<<<dim3(3 * D_ * E_ / 256), dim3(256), 0, stream>>>(Wq, Wk, Wv, Wt);

    dim3 g1(B_ * S_ / 16, 3);
    qkv_proj_kernel<<<g1, dim3(256), 0, stream>>>(x, Wt, bq, bk, bv, Qw, Kw, Vtw);

    dim3 g2(B_ * S_ / BQ);
    flash_attn_kernel<<<g2, dim3(128), 0, stream>>>(Qw, Kw, Vtw, out);
}
